// AlphaKnot_22299470200871
// MI455X (gfx1250) — matrix operand order repaired and checked by reading, not compiled, not run on hardware
//
#include <hip/hip_runtime.h>
#include <hip/hip_bf16.h>
#include <stddef.h>

// ---------------------------------------------------------------------------
// Problem constants (match reference)
// ---------------------------------------------------------------------------
constexpr int NN    = 65536;   // nodes
constexpr int BB    = 128;     // graphs
constexpr int DD    = 512;     // model dim
constexpr int HH    = 8;       // heads
constexpr int DKK   = 64;      // d_k = d_v
constexpr int DFFN  = 2048;    // ffn hidden
constexpr int LL    = 4;       // layers
constexpr int MOVES = 10;
constexpr int PDH   = 1024;    // policy hidden
constexpr int VDH   = 1024;    // value hidden
constexpr size_t ND  = (size_t)NN * DD;        // 33,554,432
constexpr size_t NH  = (size_t)NN * DFFN;      // 134,217,728
constexpr size_t NLOG = (size_t)HH * 5 * NN;   // 2,621,440 logits
constexpr size_t WMAT = (size_t)DD * DD;       // 262,144 per packed matrix
constexpr size_t WLAYER = 11 * WMAT;           // Q + 5K + 5V per layer

// GEMM epilogue flags
#define GF_RELU 1
#define GF_BIAS 2
#define GF_RES  4

typedef __attribute__((ext_vector_type(16))) __bf16 v16bf;
typedef __attribute__((ext_vector_type(2)))  __bf16 v2bf;
typedef __attribute__((ext_vector_type(8)))  float  v8f;

// fp32 pair -> packed bf16x2 via single v_cvt_pk_bf16_f32.
__device__ inline unsigned pk2(float lo, float hi) {
    v2bf v;
    v.x = (__bf16)lo;
    v.y = (__bf16)hi;
    return __builtin_bit_cast(unsigned, v);
}

// Build a 16x32 bf16 WMMA fragment from a contiguous 32-element LDS row.
// gfx1250 layout: lane-half 0 needs elements [0..7] and [16..23]; half 1
// needs [8..15] and [24..31] -> two contiguous 16-byte runs -> 2x ds_load_b128.
__device__ inline v16bf load_frag(const __bf16* __restrict__ rowp, int hlf) {
    v16bf f;
    const int base = hlf ? 8 : 0;
    #pragma unroll
    for (int j = 0; j < 8; ++j) {
        int k = base + ((j < 4) ? (2 * j) : (16 + 2 * (j - 4))); // 0,2,4,6,16,18,20,22
        f[2 * j]     = rowp[k];
        f[2 * j + 1] = rowp[k + 1];
    }
    return f;
}

// ---------------------------------------------------------------------------
// Tiled WMMA GEMM: C[M x Ncols] = act(A[M x Kdim] @ Bw[Kdim x Ncols] + bias + Res)
// Block tile 128(M) x 128(N), K-step 32, double-buffered LDS.
// 8 waves; wave w owns rows [w*16, w*16+16) x 128 cols = 8 WMMA accumulators.
// A tile row-major [m][k]; B tile transposed [n][k] so both fragment types
// load as 2x ds_load_b128.  B fragments are software-pipelined (load nt+1
// while WMMA nt executes) so LDS latency overlaps the matrix pipe.
// Optional row gather: A row m comes from row_idx[m*ridx_stride + ridx_off].
// ---------------------------------------------------------------------------
__global__ __launch_bounds__(256) void wmma_gemm(
    const float* __restrict__ A, const float* __restrict__ Bw,
    const float* __restrict__ bias, const float* __restrict__ Res,
    float* __restrict__ C,
    const int* __restrict__ row_idx, int ridx_stride, int ridx_off,
    int M, int Kdim, int Ncols, int flags)
{
    __shared__ alignas(16) unsigned sA32[2][128 * 16];  // [m][k/2] bf16-pairs
    __shared__ alignas(16) unsigned sB32[2][128 * 16];  // [n][k/2] bf16-pairs

    const int tid  = threadIdx.x;
    const int wave = tid >> 5;
    const int lane = tid & 31;
    const int hlf  = lane >> 4;       // 0: lanes 0-15, 1: lanes 16-31
    const int mrow = lane & 15;

    const int m0 = blockIdx.y * 128;
    const int n0 = blockIdx.x * 128;

    v8f zero = {};
    v8f acc[8];
    #pragma unroll
    for (int t = 0; t < 8; ++t) acc[t] = zero;

    // A staging: 16 elems/thread -> row = tid/2, pair base = (tid&1)*8
    const int arow  = tid >> 1;
    const int acol2 = (tid & 1) << 3;          // uint index {0,8}
    int srow = m0 + arow;
    if (row_idx) srow = row_idx[(size_t)srow * ridx_stride + ridx_off];
    const float* aptr = A + (size_t)srow * Kdim + (acol2 << 1);

    // B staging: 4x4 sub-block/thread -> cols bn..bn+3, k-rows bk..bk+3
    const int bn = (tid & 31) << 2;            // 0..124
    const int bk = (tid >> 5) << 2;            // 0..28
    const bool bfull = (n0 + 128 <= Ncols);

    auto stageA = [&](int buf, int k0) {
        const float4* p4 = (const float4*)(aptr + k0);
        float4 f0 = p4[0], f1 = p4[1], f2 = p4[2], f3 = p4[3];
        uint4 u0 = { pk2(f0.x, f0.y), pk2(f0.z, f0.w), pk2(f1.x, f1.y), pk2(f1.z, f1.w) };
        uint4 u1 = { pk2(f2.x, f2.y), pk2(f2.z, f2.w), pk2(f3.x, f3.y), pk2(f3.z, f3.w) };
        uint4* d = (uint4*)&sA32[buf][arow * 16 + acol2];
        d[0] = u0; d[1] = u1;
    };

    auto stageB = [&](int buf, int k0) {
        float v[4][4];
        if (bfull) {
            #pragma unroll
            for (int i = 0; i < 4; ++i) {
                float4 f = *(const float4*)(Bw + (size_t)(k0 + bk + i) * Ncols + n0 + bn);
                v[i][0] = f.x; v[i][1] = f.y; v[i][2] = f.z; v[i][3] = f.w;
            }
        } else {
            #pragma unroll
            for (int i = 0; i < 4; ++i) {
                #pragma unroll
                for (int c = 0; c < 4; ++c) {
                    int nc = n0 + bn + c;
                    v[i][c] = (nc < Ncols) ? Bw[(size_t)(k0 + bk + i) * Ncols + nc] : 0.0f;
                }
            }
        }
        #pragma unroll
        for (int c = 0; c < 4; ++c) {
            unsigned long long uv = (unsigned long long)pk2(v[0][c], v[1][c])
                                  | ((unsigned long long)pk2(v[2][c], v[3][c]) << 32);
            *(unsigned long long*)&sB32[buf][(bn + c) * 16 + (bk >> 1)] = uv;
        }
    };

    const int am = (wave << 4) + mrow;
    int buf = 0;
    stageA(0, 0);
    stageB(0, 0);

    for (int k0 = 0; k0 < Kdim; k0 += 32) {
        __syncthreads();
        if (k0 + 32 < Kdim) {       // prefetch next tile into the other buffer
            stageA(buf ^ 1, k0 + 32);
            stageB(buf ^ 1, k0 + 32);
        }
        const __bf16* pa = (const __bf16*)&sA32[buf][0];
        const __bf16* pb = (const __bf16*)&sB32[buf][0];
        v16bf af = load_frag(pa + am * 32, hlf);
        // software-pipelined B fragments: load nt+1 while WMMA nt executes
        v16bf bcur = load_frag(pb + (size_t)mrow * 32, hlf);
        #pragma unroll
        for (int nt = 0; nt < 8; ++nt) {
            v16bf bnext;
            if (nt < 7) bnext = load_frag(pb + (((nt + 1) << 4) + mrow) * 32, hlf);
            acc[nt] = __builtin_amdgcn_wmma_f32_16x16x32_bf16(
                false, af, false, bcur, (short)0, acc[nt], false, false);
            bcur = bnext;
        }
        buf ^= 1;
    }

    // Epilogue: C/D layout -> VGPR v holds M = v + 8*hlf, N = lane%16
    #pragma unroll
    for (int nt = 0; nt < 8; ++nt) {
        int n = n0 + (nt << 4) + mrow;
        if (n >= Ncols) continue;
        float bv = (flags & GF_BIAS) ? bias[n] : 0.0f;
        #pragma unroll
        for (int vr = 0; vr < 8; ++vr) {
            int m = m0 + (wave << 4) + vr + (hlf << 3);
            float v = acc[nt][vr] + bv;
            if (flags & GF_RES) v += Res[(size_t)m * Ncols + n];
            if (flags & GF_RELU) v = v > 0.0f ? v : 0.0f;
            C[(size_t)m * Ncols + n] = v;
        }
    }
}

// ---------------------------------------------------------------------------
// Pack wq/wk/wv (head-interleaved) into row-major D x (H*DK) matrices.
// Layout per layer: [Q][K_0..K_4][V_0..V_4], each 512x512.
// ---------------------------------------------------------------------------
__global__ void pack_qkv(const float* __restrict__ wq, const float* __restrict__ wk,
                         const float* __restrict__ wv, float* __restrict__ wp)
{
    size_t idx = (size_t)blockIdx.x * 256 + threadIdx.x;
    if (idx >= (size_t)LL * WLAYER) return;
    int layer = (int)(idx / WLAYER);
    size_t rem = idx % WLAYER;
    int mat = (int)(rem / WMAT);
    int e   = (int)(rem % WMAT);
    int d = e >> 9;        // 0..511
    int c = e & 511;       // h*64 + k
    int h = c >> 6, k = c & 63;
    float val;
    if (mat == 0) {
        val = wq[(((size_t)layer * HH + h) * DD + d) * DKK + k];
    } else if (mat <= 5) {
        int r = mat - 1;
        val = wk[((((size_t)layer * HH + h) * 5 + r) * DD + d) * DKK + k];
    } else {
        int r = mat - 6;
        val = wv[((((size_t)layer * HH + h) * 5 + r) * DD + d) * DKK + k];
    }
    wp[idx] = val;
}

// ---------------------------------------------------------------------------
// Input embedding: X[n,c] = b[c] + sum_{j<6} x6[n,j] * w[j,c]
// ---------------------------------------------------------------------------
__global__ void embed_k(const float* __restrict__ x6, const float* __restrict__ w,
                        const float* __restrict__ b, float* __restrict__ X)
{
    size_t idx = (size_t)blockIdx.x * 256 + threadIdx.x;
    int n = (int)(idx >> 9);
    int c = (int)(idx & 511);
    const float* xr = x6 + (size_t)n * 6;
    float s = b[c];
    #pragma unroll
    for (int j = 0; j < 6; ++j) s += xr[j] * w[j * DD + c];
    X[idx] = s;
}

// ---------------------------------------------------------------------------
// Attention logit for slot r: logits[(h*5+r)*N + n] = 0.125 * <Q[n,h], K_r[n,h]>
// ---------------------------------------------------------------------------
__global__ void attn_score(const float* __restrict__ Q, const float* __restrict__ Kr,
                           float* __restrict__ logits, int r)
{
    size_t idx = (size_t)blockIdx.x * 256 + threadIdx.x;   // n*8 + h
    int n = (int)(idx >> 3);
    int h = (int)(idx & 7);
    const float* q = Q  + (size_t)n * DD + h * DKK;
    const float* k = Kr + (size_t)n * DD + h * DKK;
    float s = 0.0f;
    #pragma unroll 8
    for (int j = 0; j < DKK; ++j) s += q[j] * k[j];
    logits[((size_t)(h * 5 + r) << 16) + n] = s * 0.125f;   // 1/sqrt(64)
}

// ---------------------------------------------------------------------------
// Softmax over the NODE axis: 40 columns of length N (reference quirk: axis=1).
// One block per (h,r) column.  In-place.
// ---------------------------------------------------------------------------
__global__ void softmax_nodes(float* __restrict__ logits)
{
    __shared__ float red[256];
    const int tid = threadIdx.x;
    float* p = logits + ((size_t)blockIdx.x << 16);

    float mx = -3.4e38f;
    for (int i = tid; i < NN; i += 256) mx = fmaxf(mx, p[i]);
    red[tid] = mx; __syncthreads();
    for (int s = 128; s > 0; s >>= 1) { if (tid < s) red[tid] = fmaxf(red[tid], red[tid + s]); __syncthreads(); }
    mx = red[0]; __syncthreads();

    float sum = 0.0f;
    for (int i = tid; i < NN; i += 256) sum += __expf(p[i] - mx);
    red[tid] = sum; __syncthreads();
    for (int s = 128; s > 0; s >>= 1) { if (tid < s) red[tid] += red[tid + s]; __syncthreads(); }
    float inv = 1.0f / red[0];

    for (int i = tid; i < NN; i += 256) p[i] = __expf(p[i] - mx) * inv;
}

// ---------------------------------------------------------------------------
// Z accumulation for slot r: Z[n, h*64+v] (+)= probs[(h*5+r)*N + n] * Vr[n, h*64+v]
// ---------------------------------------------------------------------------
__global__ void z_accum(const float* __restrict__ probs, const float* __restrict__ Vr,
                        float* __restrict__ Z, int r)
{
    size_t idx = (size_t)blockIdx.x * 256 + threadIdx.x;
    int n = (int)(idx >> 9);
    int c = (int)(idx & 511);
    int h = c >> 6;
    float pv = probs[((size_t)(h * 5 + r) << 16) + n];
    float v  = pv * Vr[idx];
    if (r == 0) Z[idx] = v; else Z[idx] += v;
}

// ---------------------------------------------------------------------------
// LayerNorm over rows of 512, optional residual: O = LN(X + R) * g + b
// One wave32 per row (16 elems/lane), 8 rows per block.
// ---------------------------------------------------------------------------
__global__ __launch_bounds__(256) void add_layernorm(
    const float* __restrict__ X, const float* __restrict__ R,
    const float* __restrict__ g, const float* __restrict__ b,
    float* __restrict__ O)
{
    const int wave = threadIdx.x >> 5;
    const int lane = threadIdx.x & 31;
    const int row  = blockIdx.x * 8 + wave;
    const float* xp = X + (size_t)row * DD;
    const float* rp = R ? (R + (size_t)row * DD) : nullptr;

    float v[16];
    float s = 0.0f;
    #pragma unroll
    for (int j = 0; j < 16; ++j) {
        float t = xp[lane + j * 32];
        if (rp) t += rp[lane + j * 32];
        v[j] = t; s += t;
    }
    #pragma unroll
    for (int m = 16; m > 0; m >>= 1) s += __shfl_xor(s, m, 32);
    const float mean = s * (1.0f / DD);

    float s2 = 0.0f;
    #pragma unroll
    for (int j = 0; j < 16; ++j) { float d = v[j] - mean; s2 += d * d; }
    #pragma unroll
    for (int m = 16; m > 0; m >>= 1) s2 += __shfl_xor(s2, m, 32);
    const float rs = rsqrtf(s2 * (1.0f / DD) + 1e-5f);

    float* op = O + (size_t)row * DD;
    #pragma unroll
    for (int j = 0; j < 16; ++j) {
        int c = lane + j * 32;
        op[c] = (v[j] - mean) * rs * g[c] + b[c];
    }
}

// ---------------------------------------------------------------------------
// Segment mean pool: graphs are contiguous 512-node blocks.
// ---------------------------------------------------------------------------
__global__ void pool_k(const float* __restrict__ X, const int* __restrict__ bs,
                       float* __restrict__ pooled)
{
    const int bq = blockIdx.x, tid = threadIdx.x;
    float cnt = (float)bs[bq];
    if (cnt < 1e-9f) cnt = 1e-9f;
    const float inv = 1.0f / cnt;
    for (int c = tid; c < DD; c += 256) {
        const float* p = X + (size_t)bq * 512 * DD + c;
        float s = 0.0f;
        for (int n2 = 0; n2 < 512; ++n2) s += p[(size_t)n2 * DD];
        pooled[bq * DD + c] = s * inv;
    }
}

// ---------------------------------------------------------------------------
// Value head: tanh(relu(pooled @ w1 + b1) @ w2 + b2).  One block per graph.
// ---------------------------------------------------------------------------
__global__ void value_head(const float* __restrict__ pooled,
                           const float* __restrict__ w1, const float* __restrict__ b1,
                           const float* __restrict__ w2, const float* __restrict__ b2,
                           float* __restrict__ out)
{
    __shared__ float pr[DD];
    __shared__ float red[256];
    const int bq = blockIdx.x, tid = threadIdx.x;
    for (int c = tid; c < DD; c += 256) pr[c] = pooled[bq * DD + c];
    __syncthreads();
    float part = 0.0f;
    for (int u = tid; u < VDH; u += 256) {
        float s = b1[u];
        for (int d = 0; d < DD; ++d) s += pr[d] * w1[(size_t)d * VDH + u];
        s = s > 0.0f ? s : 0.0f;
        part += s * w2[u];
    }
    red[tid] = part; __syncthreads();
    for (int s = 128; s > 0; s >>= 1) { if (tid < s) red[tid] += red[tid + s]; __syncthreads(); }
    if (tid == 0) out[bq] = tanhf(red[0] + b2[0]);
}

// ---------------------------------------------------------------------------
// Launch
// ---------------------------------------------------------------------------
extern "C" void kernel_launch(void* const* d_in, const int* in_sizes, int n_in,
                              void* d_out, int out_size, void* d_ws, size_t ws_size,
                              hipStream_t stream)
{
    const float* x6      = (const float*)d_in[0];
    const int*   adj     = (const int*)  d_in[1];
    const int*   bsz     = (const int*)  d_in[2];
    const float* embed_w = (const float*)d_in[3];
    const float* embed_b = (const float*)d_in[4];
    const float* wq      = (const float*)d_in[5];
    const float* wk      = (const float*)d_in[6];
    const float* wv      = (const float*)d_in[7];
    const float* ffn_w1  = (const float*)d_in[8];
    const float* ffn_b1  = (const float*)d_in[9];
    const float* ffn_w2  = (const float*)d_in[10];
    const float* ffn_b2  = (const float*)d_in[11];
    const float* ln1_g   = (const float*)d_in[12];
    const float* ln1_b   = (const float*)d_in[13];
    const float* ln2_g   = (const float*)d_in[14];
    const float* ln2_b   = (const float*)d_in[15];
    const float* pol_w1  = (const float*)d_in[16];
    const float* pol_b1  = (const float*)d_in[17];
    const float* pol_w2  = (const float*)d_in[18];
    const float* pol_b2  = (const float*)d_in[19];
    const float* val_w1  = (const float*)d_in[20];
    const float* val_b1  = (const float*)d_in[21];
    const float* val_w2  = (const float*)d_in[22];
    const float* val_b2  = (const float*)d_in[23];

    float* out = (float*)d_out;               // policy (N*10) ++ value (B)

    // workspace carve-up (floats)
    float* ws      = (float*)d_ws;
    float* x       = ws;                      // N x D
    float* y       = x + ND;                  // N x D
    float* q       = y + ND;                  // N x D
    float* kv      = q + ND;                  // N x D  (streamed K_r / V_r)
    float* z       = kv + ND;                 // N x D
    float* hbuf    = z + ND;                  // N x DFF (also policy hidden)
    float* logits  = hbuf + NH;               // 40 x N
    float* pooled  = logits + NLOG;           // B x D
    float* wpack   = pooled + (size_t)BB * DD;// 4 x 11 x 512 x 512

    const dim3 blk(256);

    // 0) weight packing + embedding
    {
        size_t tot = (size_t)LL * WLAYER;
        pack_qkv<<<dim3((unsigned)((tot + 255) / 256)), blk, 0, stream>>>(wq, wk, wv, wpack);
        embed_k<<<dim3((unsigned)(ND / 256)), blk, 0, stream>>>(x6, embed_w, embed_b, x);
    }

    const dim3 g512(DD / 128, NN / 128);      // (4, 512)
    const dim3 gffn(DFFN / 128, NN / 128);    // (16, 512)
    const dim3 gpol1(PDH / 128, NN / 128);    // (8, 512)
    const dim3 gpol2(1, NN / 128);            // (1, 512)

    for (int i = 0; i < LL; ++i) {
        const float* WQ = wpack + (size_t)i * WLAYER;

        // Q = x @ WQ
        wmma_gemm<<<g512, blk, 0, stream>>>(x, WQ, nullptr, nullptr, q,
                                            nullptr, 0, 0, NN, DD, DD, 0);
        // K_r streamed -> logits
        for (int r = 0; r < 5; ++r) {
            const float* WK = wpack + (size_t)i * WLAYER + (size_t)(1 + r) * WMAT;
            wmma_gemm<<<g512, blk, 0, stream>>>(x, WK, nullptr, nullptr, kv,
                                                (r == 0) ? nullptr : adj, 4, r - 1,
                                                NN, DD, DD, 0);
            attn_score<<<dim3((unsigned)((size_t)NN * HH / 256)), blk, 0, stream>>>(q, kv, logits, r);
        }
        // global softmax over node axis (reference's axis=1)
        softmax_nodes<<<dim3(HH * 5), blk, 0, stream>>>(logits);
        // V_r streamed -> Z
        for (int r = 0; r < 5; ++r) {
            const float* WV = wpack + (size_t)i * WLAYER + (size_t)(6 + r) * WMAT;
            wmma_gemm<<<g512, blk, 0, stream>>>(x, WV, nullptr, nullptr, kv,
                                                (r == 0) ? nullptr : adj, 4, r - 1,
                                                NN, DD, DD, 0);
            z_accum<<<dim3((unsigned)(ND / 256)), blk, 0, stream>>>(logits, kv, z, r);
        }
        // y = LN(x + Z)
        add_layernorm<<<dim3(NN / 8), blk, 0, stream>>>(z, x, ln1_g + i * DD, ln1_b + i * DD, y);
        // FFN: h = relu(y @ w1 + b1); z = h @ w2 + b2 + y; x = LN(z)
        wmma_gemm<<<gffn, blk, 0, stream>>>(y, ffn_w1 + (size_t)i * DD * DFFN,
                                            ffn_b1 + (size_t)i * DFFN, nullptr, hbuf,
                                            nullptr, 0, 0, NN, DD, DFFN, GF_BIAS | GF_RELU);
        wmma_gemm<<<g512, blk, 0, stream>>>(hbuf, ffn_w2 + (size_t)i * DFFN * DD,
                                            ffn_b2 + (size_t)i * DD, y, z,
                                            nullptr, 0, 0, NN, DFFN, DD, GF_BIAS | GF_RES);
        add_layernorm<<<dim3(NN / 8), blk, 0, stream>>>(z, nullptr, ln2_g + i * DD, ln2_b + i * DD, x);
    }

    // policy head -> out[0 : N*MOVES]
    wmma_gemm<<<gpol1, blk, 0, stream>>>(x, pol_w1, pol_b1, nullptr, hbuf,
                                         nullptr, 0, 0, NN, DD, PDH, GF_BIAS | GF_RELU);
    wmma_gemm<<<gpol2, blk, 0, stream>>>(hbuf, pol_w2, pol_b2, nullptr, out,
                                         nullptr, 0, 0, NN, PDH, MOVES, GF_BIAS);

    // value head -> out[N*MOVES : N*MOVES + B]
    pool_k<<<dim3(BB), blk, 0, stream>>>(x, bsz, pooled);
    value_head<<<dim3(BB), blk, 0, stream>>>(pooled, val_w1, val_b1, val_w2, val_b2,
                                             out + (size_t)NN * MOVES);
}
